// AttentionLayer_10273561772576
// MI455X (gfx1250) — compile-verified, matrix operand order checked by reading, not run on hardware
//
#include <hip/hip_runtime.h>
#include <hip/hip_bf16.h>
#include <stdint.h>

// ---------------------------------------------------------------------------
// Flash-attention style kernel for MI455X (gfx1250, wave32, WMMA bf16).
//   scores = Q @ V^T ; P = softmax(scores) ; O = P @ V   (f32 I/O)
// B=8, SQ=SV=2048, D=1024. Compute in bf16 WMMA 16x16x32, f32 accumulate.
// ---------------------------------------------------------------------------

typedef __attribute__((ext_vector_type(16))) __bf16 v16bf;
typedef __attribute__((ext_vector_type(8)))  float  v8f;

#define BATCH 8
#define SQ    2048
#define SV    2048
#define HD    1024
#define MTILE 32          // query rows per workgroup
#define KT    32          // kv rows per inner tile
#define NKV   (SV / KT)   // 64 kv tiles

#define QSTR  1032        // Vs row stride (bf16 elems): 2064B, 16B aligned
#define VSTR  40          // VsT row stride: 80B, 16B aligned
#define PSTR  40          // Ps row stride
#define SSTR  33          // score-partial row stride (f32)

struct SMem {
  __bf16 Vs [KT][QSTR];       // V tile, row-major (kv-row, d)      66,048 B
  __bf16 VsT[HD][VSTR];       // V tile, transposed (d, kv-row)     81,920 B
  float  Sp[2][MTILE][SSTR];  // K-split score partials              8,448 B
  __bf16 Ps [MTILE][PSTR];    // softmaxed P tile (bf16)             2,560 B
  float  m_row[MTILE];
  float  l_row[MTILE];
  float  a_row[MTILE];        // per-iteration rescale alpha
};                            // total ~159 KB < 320 KB/WGP

// Load one 16x32 (A) or 32x16-as-rows (B) bf16 fragment from LDS following
// the CDNA5 16-bit matrix VGPR layout:
//   lanes 0-15: VGPR0-3 = K0..7, VGPR4-7 = K16..23 ; lanes 16-31: +8 on K.
// => two 16-byte DS loads per lane.
__device__ __forceinline__ v16bf lds_frag(const __bf16* base, int stride,
                                          int row, int kbase) {
  const int lane = threadIdx.x & 31;
  const int r  = row + (lane & 15);
  const int k0 = kbase + ((lane >> 4) << 3);
  const __bf16* p = base + r * stride + k0;
  union { uint4 q[2]; v16bf v; } u;
  u.q[0] = *(const uint4*)(p);
  u.q[1] = *(const uint4*)(p + 16);
  return u.v;
}

// Build the same fragment directly from global f32 (used once for Q).
__device__ __forceinline__ v16bf cvt_frag_global(const float* p) {
  float4 f0 = *(const float4*)(p);
  float4 f1 = *(const float4*)(p + 4);
  float4 f2 = *(const float4*)(p + 16);
  float4 f3 = *(const float4*)(p + 20);
  v16bf v;
  v[0]  = (__bf16)f0.x; v[1]  = (__bf16)f0.y; v[2]  = (__bf16)f0.z; v[3]  = (__bf16)f0.w;
  v[4]  = (__bf16)f1.x; v[5]  = (__bf16)f1.y; v[6]  = (__bf16)f1.z; v[7]  = (__bf16)f1.w;
  v[8]  = (__bf16)f2.x; v[9]  = (__bf16)f2.y; v[10] = (__bf16)f2.z; v[11] = (__bf16)f2.w;
  v[12] = (__bf16)f3.x; v[13] = (__bf16)f3.y; v[14] = (__bf16)f3.z; v[15] = (__bf16)f3.w;
  return v;
}

__global__ __launch_bounds__(256, 1)
void AttentionLayer_flash_wmma(const float* __restrict__ Q,
                               const float* __restrict__ V,
                               float* __restrict__ out) {
  __shared__ __align__(16) SMem sm;

  const int tid  = threadIdx.x;
  const int lane = tid & 31;
  const int w    = tid >> 5;       // wave id 0..7
  const int half = lane >> 4;      // 0 = lanes 0-15, 1 = lanes 16-31
  const int ln   = lane & 15;

  const int qtile = blockIdx.x;    // 0..63
  const int bi    = blockIdx.y;    // 0..7

  // GEMM1 role: waves (ks=w>>2) split K=1024 in halves; tile (tr,tc) = w&3.
  const int ks = w >> 2;
  const int tr = (w & 3) >> 1;
  const int tc = (w & 3) & 1;
  // GEMM2 / output role: wave owns rows [16*rblk,+16), cols [256*cg,+256).
  const int rblk = w >> 2;
  const int cg   = w & 3;

  if (tid < MTILE) { sm.m_row[tid] = -__builtin_inff(); sm.l_row[tid] = 0.0f; }

  // ---- Preload Q fragments (reused for all 64 kv tiles) -------------------
  const float* Qrow =
      Q + ((size_t)(bi * SQ + qtile * MTILE + tr * 16 + ln)) * HD;
  v16bf qfrag[16];
#pragma unroll
  for (int kk = 0; kk < 16; ++kk)
    qfrag[kk] = cvt_frag_global(Qrow + ks * 512 + kk * 32 + (half << 3));

  v8f acc[16];
#pragma unroll
  for (int j = 0; j < 16; ++j) acc[j] = (v8f){};

  const float* Vbase = V + (size_t)bi * SV * HD;

  for (int kv = 0; kv < NKV; ++kv) {
    __syncthreads();  // previous iteration done with Vs/VsT/Ps

    // ---- Stage V tile into LDS (bf16, both orientations) ------------------
    const float* vsrc = Vbase + (size_t)kv * KT * HD;
#pragma unroll 4
    for (int r = 0; r < KT; ++r) {
      const float4 f = *(const float4*)(vsrc + r * HD + tid * 4);
      if (kv + 1 < NKV && (tid & 3) == 0)
        __builtin_prefetch(vsrc + (size_t)KT * HD + r * HD + tid * 4, 0, 1);
      __bf16 h0 = (__bf16)f.x, h1 = (__bf16)f.y,
             h2 = (__bf16)f.z, h3 = (__bf16)f.w;
      union { __bf16 h[4]; uint2 u; } pk;
      pk.h[0] = h0; pk.h[1] = h1; pk.h[2] = h2; pk.h[3] = h3;
      *(uint2*)&sm.Vs[r][tid * 4] = pk.u;
      const int d0 = tid * 4;
      sm.VsT[d0 + 0][r] = h0; sm.VsT[d0 + 1][r] = h1;
      sm.VsT[d0 + 2][r] = h2; sm.VsT[d0 + 3][r] = h3;
    }
    __syncthreads();

    // ---- GEMM1: S_partial(tr,tc) = Q[tr] . V[tc]^T over K half `ks` -------
    v8f s = (v8f){};
#pragma unroll
    for (int kk = 0; kk < 16; ++kk) {
      v16bf bv = lds_frag(&sm.Vs[0][0], QSTR, tc * 16, ks * 512 + kk * 32);
      s = __builtin_amdgcn_wmma_f32_16x16x32_bf16(
              false, qfrag[kk], false, bv, (short)0, s, false, false);
    }
#pragma unroll
    for (int i = 0; i < 8; ++i)
      sm.Sp[ks][tr * 16 + i + 8 * half][tc * 16 + ln] = s[i];
    __syncthreads();

    // ---- Online softmax (8 lanes per row, shuffle reductions) -------------
    {
      const int row = tid >> 3, kg = tid & 7, c0 = kg * 4;
      float sv[4];
      float mx = -__builtin_inff();
#pragma unroll
      for (int u = 0; u < 4; ++u) {
        sv[u] = sm.Sp[0][row][c0 + u] + sm.Sp[1][row][c0 + u];
        mx = fmaxf(mx, sv[u]);
      }
      mx = fmaxf(mx, __shfl_xor(mx, 1));
      mx = fmaxf(mx, __shfl_xor(mx, 2));
      mx = fmaxf(mx, __shfl_xor(mx, 4));
      const float m_old = sm.m_row[row];
      const float m_new = fmaxf(m_old, mx);
      const float alpha = __expf(m_old - m_new);
      float psum = 0.0f;
#pragma unroll
      for (int u = 0; u < 4; ++u) {
        const float p = __expf(sv[u] - m_new);
        psum += p;
        sm.Ps[row][c0 + u] = (__bf16)p;
      }
      psum += __shfl_xor(psum, 1);
      psum += __shfl_xor(psum, 2);
      psum += __shfl_xor(psum, 4);
      if (kg == 0) {
        sm.m_row[row] = m_new;
        sm.l_row[row] = sm.l_row[row] * alpha + psum;
        sm.a_row[row] = alpha;
      }
    }
    __syncthreads();

    // ---- GEMM2: O[rblk, cg*256..] = alpha*O + P[rblk] @ V_tile ------------
    {
      const v16bf aP = lds_frag(&sm.Ps[0][0], PSTR, rblk * 16, 0);
      float alph[8];
#pragma unroll
      for (int i = 0; i < 8; ++i)
        alph[i] = sm.a_row[rblk * 16 + i + 8 * half];
#pragma unroll
      for (int j = 0; j < 16; ++j) {
        v16bf bv = lds_frag(&sm.VsT[0][0], VSTR, cg * 256 + j * 16, 0);
#pragma unroll
        for (int i = 0; i < 8; ++i) acc[j][i] *= alph[i];
        acc[j] = __builtin_amdgcn_wmma_f32_16x16x32_bf16(
                     false, aP, false, bv, (short)0, acc[j], false, false);
      }
    }
  }

  // ---- Epilogue: divide by softmax denominator, store f32 ------------------
  float linv[8];
#pragma unroll
  for (int i = 0; i < 8; ++i)
    linv[i] = 1.0f / sm.l_row[rblk * 16 + i + 8 * half];
  float* obase = out + ((size_t)(bi * SQ + qtile * MTILE + rblk * 16)) * HD;
#pragma unroll
  for (int j = 0; j < 16; ++j) {
    const int d = cg * 256 + j * 16 + ln;
#pragma unroll
    for (int i = 0; i < 8; ++i)
      obase[(size_t)(i + 8 * half) * HD + d] = acc[j][i] * linv[i];
  }
}

extern "C" void kernel_launch(void* const* d_in, const int* in_sizes, int n_in,
                              void* d_out, int out_size, void* d_ws,
                              size_t ws_size, hipStream_t stream) {
  (void)in_sizes; (void)n_in; (void)out_size; (void)d_ws; (void)ws_size;
  const float* Q = (const float*)d_in[0];
  const float* V = (const float*)d_in[1];
  float* O = (float*)d_out;
  dim3 grid(SQ / MTILE, BATCH);   // 64 x 8 workgroups
  AttentionLayer_flash_wmma<<<grid, 256, 0, stream>>>(Q, V, O);
}
